// myRNN_10488310137356
// MI455X (gfx1250) — matrix operand order repaired and checked by reading, not compiled, not run on hardware
//
#include <hip/hip_runtime.h>

#define HID    100
#define TSEQ   10
#define NB     5
#define WAVES  8
#define ROWS   16
#define KP     128
#define WROWS  312
#define GJT    7
#define VPITCH 15

typedef _Float16 v16h __attribute__((ext_vector_type(16)));
typedef float    v8f  __attribute__((ext_vector_type(8)));
typedef float    f4   __attribute__((ext_vector_type(4)));

union FragU { v16h v; struct { f4 lo, hi; } q; };

__device__ __forceinline__ v8f wmma_f16(v16h a, v16h b, v8f c) {
  // D = A(16x32 f16) * B(32x16 f16) + C(16x16 f32)
  return __builtin_amdgcn_wmma_f32_16x16x32_f16(false, a, false, b, (short)0, c, false, false);
}

// A fragment: lane = M (batch row), lanes 0..15 K=kbase..kbase+15,
// lanes 16..31 K=kbase+16..kbase+31, contiguous per lane. Two 16B LDS loads.
__device__ __forceinline__ v16h load_afrag(const _Float16* h, int lane, int kbase) {
  const int m  = lane & 15;
  const int k0 = kbase + (lane & 16);          // hi half-wave -> +16
  FragU u;
  u.q.lo = *(const f4*)(h + m * KP + k0);
  u.q.hi = *(const f4*)(h + m * KP + k0 + 8);
  return u.v;
}

// B fragment: lane = N (gate col within tile), lanes 0..15 K=kbase..kbase+15,
// lanes 16..31 K=kbase+16..kbase+31, contiguous per lane. Two 16B LDS loads.
__device__ __forceinline__ v16h load_bfrag(const _Float16* w, int rowbase, int lane, int kbase) {
  const int r  = rowbase + (lane & 15);
  const int k0 = kbase + (lane & 16);
  FragU u;
  u.q.lo = *(const f4*)(w + r * KP + k0);
  u.q.hi = *(const f4*)(w + r * KP + k0 + 8);
  return u.v;
}

// Store a 16x16 f32 C/D tile (lane = col, vgpr = row) to f16 LDS [m][KP]
__device__ __forceinline__ void store_htile(_Float16* h, int lane, int jt, v8f d) {
  const int j  = (jt << 4) + (lane & 15);
  const int mb = (lane & 16) >> 1;             // rows 8..15 on upper half-wave
  #pragma unroll
  for (int e = 0; e < 8; ++e) h[(mb + e) * KP + j] = (_Float16)d[e];
}

__device__ __forceinline__ float sigm(float x)      { return 1.0f / (1.0f + __expf(-x)); }
__device__ __forceinline__ float tanh_fast(float x) { return 1.0f - 2.0f / (__expf(2.0f * x) + 1.0f); }

__global__ void __launch_bounds__(WAVES * 32) gru5_wmma(
    const float* __restrict__ x,
    const float* __restrict__ Wih0, const float* __restrict__ Whh0,
    const float* __restrict__ bih0, const float* __restrict__ bhh0,
    const float* __restrict__ Wih1, const float* __restrict__ Whh1,
    const float* __restrict__ bih1, const float* __restrict__ bhh1,
    const float* __restrict__ Wlin, const float* __restrict__ blin,
    float* __restrict__ out)
{
  // Resident f16 weights in LDS (zero padded rows >=300 and cols >=100)
  __shared__ __align__(16) _Float16 sW[3][WROWS * KP];          // Whh0, Wih1, Whh1
  __shared__ __align__(16) _Float16 sH[2 * WAVES][ROWS * KP];   // per-wave h state mirrors (f16)
  __shared__ float sVals[WAVES][ROWS * VPITCH];                 // sliding input window
  __shared__ float sWi0[WROWS];
  __shared__ float sB[4][WROWS];                                // bih0,bhh0,bih1,bhh1 (padded)
  __shared__ float sWl[GJT * 16];

  const int tid = threadIdx.x;
  const int nth = WAVES * 32;

  { // zero all padded LDS regions
    uint32_t* p = (uint32_t*)&sW[0][0];
    for (int i = tid; i < 3 * WROWS * KP / 2; i += nth) p[i] = 0u;
    uint32_t* q = (uint32_t*)&sH[0][0];
    for (int i = tid; i < 2 * WAVES * ROWS * KP / 2; i += nth) q[i] = 0u;
  }
  __syncthreads();
  // fill weights f32 -> f16
  for (int i = tid; i < 3 * HID * HID; i += nth) {
    const int nr = i / HID, k = i % HID;
    sW[0][nr * KP + k] = (_Float16)Whh0[i];
    sW[1][nr * KP + k] = (_Float16)Wih1[i];
    sW[2][nr * KP + k] = (_Float16)Whh1[i];
  }
  for (int i = tid; i < WROWS; i += nth) {
    const bool v = (i < 3 * HID);
    sWi0[i]  = v ? Wih0[i] : 0.0f;
    sB[0][i] = v ? bih0[i] : 0.0f;
    sB[1][i] = v ? bhh0[i] : 0.0f;
    sB[2][i] = v ? bih1[i] : 0.0f;
    sB[3][i] = v ? bhh1[i] : 0.0f;
  }
  for (int i = tid; i < GJT * 16; i += nth) sWl[i] = (i < HID) ? Wlin[i] : 0.0f;

  const int wave = tid >> 5;
  const int lane = tid & 31;
  const int bm   = (blockIdx.x * WAVES + wave) * ROWS;   // batch row base of this wave
  _Float16* h0buf = &sH[wave][0];
  _Float16* h1buf = &sH[WAVES + wave][0];
  float* vals = &sVals[wave][0];

  for (int i = lane; i < ROWS * TSEQ; i += 32) {
    const int m = i / TSEQ, t = i % TSEQ;
    vals[m * VPITCH + t] = x[(size_t)(bm + m) * TSEQ + t];
  }
  __syncthreads();

  const int jl = lane & 15;
  const int mb = (lane & 16) >> 1;
  const float blv = blin[0];

  v8f h0r[GJT], h1r[GJT];

  for (int it = 0; it < NB; ++it) {
    // h := 0 at start of each GRU pass, publish zeros
    #pragma unroll
    for (int jt = 0; jt < GJT; ++jt) {
      v8f z = {0.f,0.f,0.f,0.f,0.f,0.f,0.f,0.f};
      h0r[jt] = z; h1r[jt] = z;
      store_htile(h0buf, lane, jt, z);
      store_htile(h1buf, lane, jt, z);
    }
    asm volatile("s_wait_dscnt 0x0" ::: "memory");

    for (int t = 0; t < TSEQ; ++t) {
      float xv[8];
      #pragma unroll
      for (int e = 0; e < 8; ++e) xv[e] = vals[(mb + e) * VPITCH + (it + t)];

      // ============ layer 0 (input dim 1: gi is an outer product) ============
      for (int jt = 0; jt < GJT; ++jt) {
        const int jg = (jt << 4) + jl;
        const float wr = sWi0[jg], wz = sWi0[HID + jg], wn = sWi0[2 * HID + jg];
        const float br  = sB[0][jg] + sB[1][jg];
        const float bz  = sB[0][HID + jg] + sB[1][HID + jg];
        const float bni = sB[0][2 * HID + jg];
        const float bnh = sB[1][2 * HID + jg];
        v8f cr, cz, ch; float inn[8];
        #pragma unroll
        for (int e = 0; e < 8; ++e) {
          cr[e]  = xv[e] * wr + br;
          cz[e]  = xv[e] * wz + bz;
          ch[e]  = bnh;
          inn[e] = xv[e] * wn + bni;
        }
        #pragma unroll
        for (int kc = 0; kc < 4; ++kc) {
          const int kb = kc << 5;
          const v16h a = load_afrag(h0buf, lane, kb);
          cr = wmma_f16(a, load_bfrag(sW[0], (jt << 4),           lane, kb), cr);
          cz = wmma_f16(a, load_bfrag(sW[0], HID + (jt << 4),     lane, kb), cz);
          ch = wmma_f16(a, load_bfrag(sW[0], 2 * HID + (jt << 4), lane, kb), ch);
        }
        #pragma unroll
        for (int e = 0; e < 8; ++e) {
          const float r  = sigm(cr[e]);
          const float zg = sigm(cz[e]);
          const float n  = tanh_fast(inn[e] + r * ch[e]);
          h0r[jt][e] = (1.0f - zg) * n + zg * h0r[jt][e];
        }
      }
      #pragma unroll
      for (int jt = 0; jt < GJT; ++jt) store_htile(h0buf, lane, jt, h0r[jt]);
      asm volatile("s_wait_dscnt 0x0" ::: "memory");

      // ============ layer 1 (gi from h1_t, gh from h2_{t-1}) ============
      for (int jt = 0; jt < GJT; ++jt) {
        const int jg = (jt << 4) + jl;
        v8f cr, cz, ci, ch;
        #pragma unroll
        for (int e = 0; e < 8; ++e) {
          cr[e] = sB[2][jg] + sB[3][jg];
          cz[e] = sB[2][HID + jg] + sB[3][HID + jg];
          ci[e] = sB[2][2 * HID + jg];
          ch[e] = sB[3][2 * HID + jg];
        }
        #pragma unroll
        for (int kc = 0; kc < 4; ++kc) {
          const int kb = kc << 5;
          const v16h a1 = load_afrag(h0buf, lane, kb);
          const v16h a2 = load_afrag(h1buf, lane, kb);
          cr = wmma_f16(a1, load_bfrag(sW[1], (jt << 4),           lane, kb), cr);
          cr = wmma_f16(a2, load_bfrag(sW[2], (jt << 4),           lane, kb), cr);
          cz = wmma_f16(a1, load_bfrag(sW[1], HID + (jt << 4),     lane, kb), cz);
          cz = wmma_f16(a2, load_bfrag(sW[2], HID + (jt << 4),     lane, kb), cz);
          ci = wmma_f16(a1, load_bfrag(sW[1], 2 * HID + (jt << 4), lane, kb), ci);
          ch = wmma_f16(a2, load_bfrag(sW[2], 2 * HID + (jt << 4), lane, kb), ch);
        }
        #pragma unroll
        for (int e = 0; e < 8; ++e) {
          const float r  = sigm(cr[e]);
          const float zg = sigm(cz[e]);
          const float n  = tanh_fast(ci[e] + r * ch[e]);
          h1r[jt][e] = (1.0f - zg) * n + zg * h1r[jt][e];
        }
      }
      #pragma unroll
      for (int jt = 0; jt < GJT; ++jt) store_htile(h1buf, lane, jt, h1r[jt]);
      asm volatile("s_wait_dscnt 0x0" ::: "memory");
    } // t

    // ============ linear head: ret = h2_T @ Wlin^T + b ============
    float part[8];
    #pragma unroll
    for (int e = 0; e < 8; ++e) part[e] = 0.0f;
    #pragma unroll
    for (int jt = 0; jt < GJT; ++jt) {
      const float wl = sWl[(jt << 4) + jl];
      #pragma unroll
      for (int e = 0; e < 8; ++e) part[e] += h1r[jt][e] * wl;
    }
    #pragma unroll
    for (int off = 8; off >= 1; off >>= 1) {
      #pragma unroll
      for (int e = 0; e < 8; ++e) part[e] += __shfl_xor(part[e], off, 32);
    }
    if (jl == 0) { // one writer per half-wave: rows mb..mb+7
      #pragma unroll
      for (int e = 0; e < 8; ++e) vals[(mb + e) * VPITCH + (TSEQ + it)] = part[e] + blv;
    }
    asm volatile("s_wait_dscnt 0x0" ::: "memory");
  } // it

  if (lane < ROWS) {
    #pragma unroll
    for (int o = 0; o < NB; ++o)
      out[(size_t)(bm + lane) * NB + o] = vals[lane * VPITCH + (TSEQ + o)];
  }
}

extern "C" void kernel_launch(void* const* d_in, const int* in_sizes, int n_in,
                              void* d_out, int out_size, void* d_ws, size_t ws_size,
                              hipStream_t stream) {
  (void)n_in; (void)out_size; (void)d_ws; (void)ws_size;
  const float* x    = (const float*)d_in[0];
  const float* Wih0 = (const float*)d_in[1];
  const float* Whh0 = (const float*)d_in[2];
  const float* bih0 = (const float*)d_in[3];
  const float* bhh0 = (const float*)d_in[4];
  const float* Wih1 = (const float*)d_in[5];
  const float* Whh1 = (const float*)d_in[6];
  const float* bih1 = (const float*)d_in[7];
  const float* bhh1 = (const float*)d_in[8];
  const float* Wlin = (const float*)d_in[9];
  const float* blin = (const float*)d_in[10];
  float* out = (float*)d_out;

  const int B    = in_sizes[0] / TSEQ;          // 65536
  const int grid = B / (WAVES * ROWS);          // 512 workgroups of 256 threads
  gru5_wmma<<<grid, WAVES * 32, 0, stream>>>(x, Wih0, Whh0, bih0, bhh0,
                                             Wih1, Whh1, bih1, bhh1, Wlin, blin, out);
}
